// GAT_LSTM_50259707298012
// MI455X (gfx1250) — matrix operand order repaired and checked by reading, not compiled, not run on hardware
//
#include <hip/hip_runtime.h>

// ---------------------------------------------------------------------------
// GAT + dual-LSTM + FC, fused for MI455X (gfx1250, wave32, WMMA bf16).
//
// Pipeline:  attn = GAT(embedding, W, a, adj)   (tiny, prep kernels)
//            xm[b,t,:] = attn @ x[b,t,:]        (WMMA, fused)
//            h,c LSTM over t (recent t<4 / period t>=4)  (WMMA + VALU, fused)
//            out = [hr|hp] @ fcW^T + fcb        (WMMA, fused)
//
// All weights converted to a padded bf16 blob in d_ws whose byte layout is
// identical to the fused kernel's dynamic-LDS image (one flat copy per block).
// ---------------------------------------------------------------------------

#define NNODES 156
#define NFEAT  256
#define NHID   128
#define BATCH  16384
#define TSTEPS 24
#define HL     32      // LSTM hidden; gates = 4*HL = 128
#define NP     160     // nodes padded to multiple of 32 (K for WMMA)
#define WAVES  4       // waves per block; each wave owns a 16-row batch tile

// ---- weight blob layout (bytes); same in d_ws and in dynamic LDS ----
#define OFF_ATTN_R 0u          // bf16 [160][160]  (row i = out node, col j = K)
#define OFF_ATTN_P 51200u      // bf16 [160][160]
#define OFF_WIH_R  102400u     // bf16 [128][160]  (row n = gate, col k = node)
#define OFF_WIH_P  143360u
#define OFF_WHH_R  184320u     // bf16 [128][32]
#define OFF_WHH_P  192512u
#define OFF_FCW    200704u     // bf16 [160][64]
#define OFF_BIAS_R 221184u     // f32  [128] = bih+bhh
#define OFF_BIAS_P 221696u     // f32  [128]
#define OFF_FCB    222208u     // f32  [160]
#define WBLOB      222848u     // total blob bytes (16B multiple)
#define OFF_SBUF   222848u     // f32 scratch: s1r,s2r,s1p,s2p (4*156)

typedef __attribute__((ext_vector_type(16))) __bf16 v16bf;
typedef __attribute__((ext_vector_type(2)))  __bf16 bf16x2;
typedef __attribute__((ext_vector_type(8)))  float  v8f;

union ABf { v16bf v; unsigned u[8]; };

// HW f32 -> bf16 (single): let clang pick the native cvt on gfx1250.
__device__ __forceinline__ unsigned short f2bf_h(float x) {
  __bf16 b = (__bf16)x;
  return __builtin_bit_cast(unsigned short, b);
}

// HW f32x2 -> packed bf16x2 in one u32 (v_cvt_pk path when available).
__device__ __forceinline__ unsigned pack2_bf16(float lo, float hi) {
  bf16x2 t;
  t[0] = (__bf16)lo;
  t[1] = (__bf16)hi;
  return __builtin_bit_cast(unsigned, t);
}

// Aligned 4-byte pair load (typeless -> no TBAA hazard vs u16 stores; lowers
// to a single ds_load_b32 / coalescable b128).
__device__ __forceinline__ unsigned ld_pair(const unsigned short* p) {
  unsigned u;
  __builtin_memcpy(&u, __builtin_assume_aligned(p, 4), 4);
  return u;
}

__device__ __forceinline__ float sigm(float x) {
  return 1.0f / (1.0f + __expf(-x));
}

__device__ __forceinline__ v8f wmma_bf16(const ABf& a, const ABf& b, v8f c) {
  // D = A(16x32 bf16) * B(32x16 bf16) + C(16x16 f32)
  return __builtin_amdgcn_wmma_f32_16x16x32_bf16(
      false, a.v, false, b.v, (short)0, c, false, false);
}

// B tile from LDS stored [N][K] row-major (K-pairs contiguous -> b32 loads).
// Layout: lane n = lane&15; reg v = K pair a_k(v, lane>>4) (table order, matches A).
__device__ __forceinline__ ABf load_B(const unsigned short* __restrict__ base,
                                      int ks, int n0, int k0, int lane) {
  ABf b;
  int n  = n0 + (lane & 15);
  int kb = k0 + ((lane >> 4) << 3);
  const unsigned short* p = base + n * ks + kb;
#pragma unroll
  for (int v = 0; v < 8; ++v) b.u[v] = ld_pair(p + ((v & 4) << 2) + ((v & 3) << 1));
  return b;
}

// A-layout K index for reg v, half-wave ag (16-bit A 16x32, ISA 7.12.2)
__device__ __forceinline__ int a_k(int v, int ag) {
  return ((v & 4) << 2) | (ag << 3) | ((v & 3) << 1);
}

// ---------------------------------------------------------------------------
// Prep 1: GAT scores  s1[i] = (emb@W@a_lo)[i],  s2[i] = (emb@W@a_hi)[i]
// grid (156, 2), block 128
// ---------------------------------------------------------------------------
__global__ void gat_scores_kernel(const float* __restrict__ emb,
                                  const float* __restrict__ Wr,
                                  const float* __restrict__ ar,
                                  const float* __restrict__ Wp,
                                  const float* __restrict__ ap,
                                  float* __restrict__ sbuf) {
  int i = blockIdx.x, br = blockIdx.y, f = threadIdx.x;
  const float* W = br ? Wp : Wr;
  const float* a = br ? ap : ar;
  const float* er = emb + i * NFEAT;
  float h = 0.f;
  for (int k = 0; k < NFEAT; ++k) h = fmaf(er[k], W[k * NHID + f], h);
  __shared__ float r1[128], r2[128];
  r1[f] = h * a[f];
  r2[f] = h * a[NHID + f];
  __syncthreads();
  for (int s = 64; s > 0; s >>= 1) {
    if (f < s) { r1[f] += r1[f + s]; r2[f] += r2[f + s]; }
    __syncthreads();
  }
  if (f == 0) {
    sbuf[(br * 2 + 0) * NNODES + i] = r1[0];
    sbuf[(br * 2 + 1) * NNODES + i] = r2[0];
  }
}

// ---------------------------------------------------------------------------
// Prep 2: masked leaky-relu softmax row -> bf16 attn row (padded to 160)
// grid (156, 2), block 256
// ---------------------------------------------------------------------------
__global__ void gat_softmax_kernel(const int* __restrict__ adj,
                                   const float* __restrict__ sbuf,
                                   unsigned char* __restrict__ ws) {
  int i = blockIdx.x, br = blockIdx.y, j = threadIdx.x;
  __shared__ float r[256];
  float s1 = sbuf[(br * 2 + 0) * NNODES + i];
  float e = -3.0e38f;
  if (j < NNODES) {
    float z = s1 + sbuf[(br * 2 + 1) * NNODES + j];
    z = z > 0.f ? z : 0.2f * z;                       // leaky relu, alpha 0.2
    e = (adj[i * NNODES + j] > 0) ? z : -9.0e15f;
  }
  r[j] = e; __syncthreads();
  for (int s = 128; s > 0; s >>= 1) {
    if (j < s) r[j] = fmaxf(r[j], r[j + s]);
    __syncthreads();
  }
  float m = r[0]; __syncthreads();
  float p = (j < NNODES) ? __expf(e - m) : 0.f;
  r[j] = p; __syncthreads();
  for (int s = 128; s > 0; s >>= 1) {
    if (j < s) r[j] += r[j + s];
    __syncthreads();
  }
  float inv = 1.0f / r[0];
  unsigned short* row =
      (unsigned short*)(ws + (br ? OFF_ATTN_P : OFF_ATTN_R)) + i * NP;
  if (j < NP) row[j] = (j < NNODES) ? f2bf_h(p * inv) : (unsigned short)0;
}

// ---------------------------------------------------------------------------
// Prep 3: convert/pad all LSTM + FC weights into the bf16 blob
// grid (80, 9), block 256
// ---------------------------------------------------------------------------
__global__ void prep_weights_kernel(const float* __restrict__ wih_r,
                                    const float* __restrict__ whh_r,
                                    const float* __restrict__ bih_r,
                                    const float* __restrict__ bhh_r,
                                    const float* __restrict__ wih_p,
                                    const float* __restrict__ whh_p,
                                    const float* __restrict__ bih_p,
                                    const float* __restrict__ bhh_p,
                                    const float* __restrict__ fcW,
                                    const float* __restrict__ fcB,
                                    unsigned char* __restrict__ ws) {
  int seg = blockIdx.y;
  int idx = blockIdx.x * blockDim.x + threadIdx.x;
  switch (seg) {
    case 0: case 1: {                               // Wih: [128][156]->[128][160]
      if (idx >= 4 * HL * NP) return;
      const float* src = seg ? wih_p : wih_r;
      unsigned short* dst = (unsigned short*)(ws + (seg ? OFF_WIH_P : OFF_WIH_R));
      int n = idx / NP, k = idx % NP;
      dst[idx] = (k < NNODES) ? f2bf_h(src[n * NNODES + k]) : (unsigned short)0;
    } break;
    case 2: case 3: {                               // Whh: [128][32] dense
      if (idx >= 4 * HL * HL) return;
      const float* src = (seg == 3) ? whh_p : whh_r;
      unsigned short* dst =
          (unsigned short*)(ws + ((seg == 3) ? OFF_WHH_P : OFF_WHH_R));
      dst[idx] = f2bf_h(src[idx]);
    } break;
    case 4: {                                       // fcW: [156][64]->[160][64]
      if (idx >= NP * 2 * HL) return;
      unsigned short* dst = (unsigned short*)(ws + OFF_FCW);
      int n = idx / (2 * HL), k = idx % (2 * HL);
      dst[idx] = (n < NNODES) ? f2bf_h(fcW[n * 2 * HL + k]) : (unsigned short)0;
    } break;
    case 5: case 6: {                               // bias = bih + bhh (f32)
      if (idx >= 4 * HL) return;
      const float* b1 = (seg == 6) ? bih_p : bih_r;
      const float* b2 = (seg == 6) ? bhh_p : bhh_r;
      float* dst = (float*)(ws + ((seg == 6) ? OFF_BIAS_P : OFF_BIAS_R));
      dst[idx] = b1[idx] + b2[idx];
    } break;
    case 7: {                                       // fcb padded (f32)
      if (idx >= NP) return;
      float* dst = (float*)(ws + OFF_FCB);
      dst[idx] = (idx < NNODES) ? fcB[idx] : 0.f;
    } break;
    case 8: {                                       // attn pad rows 156..159 = 0
      if (idx >= 2 * 4 * NP) return;
      int br = idx / (4 * NP), rem = idx % (4 * NP);
      unsigned short* dst =
          (unsigned short*)(ws + (br ? OFF_ATTN_P : OFF_ATTN_R));
      dst[NNODES * NP + rem] = 0;
    } break;
  }
}

// ---------------------------------------------------------------------------
// Fused: attn-mix GEMM -> LSTM (24 steps) -> FC, one 16-row tile per wave.
// grid 256, block 128 (4 waves), dynamic LDS = WBLOB bytes (weight image).
// ---------------------------------------------------------------------------
__global__ __launch_bounds__(128, 1)
void fused_gat_lstm_kernel(const float* __restrict__ x,
                           const unsigned char* __restrict__ wsblob,
                           float* __restrict__ out) {
  extern __shared__ unsigned char smem[];                       // weight image
  __shared__ __align__(16) unsigned short xmix_st[WAVES][16 * NP];
  __shared__ __align__(16) unsigned short h_st[WAVES][16 * HL];

  // ---- pull weight blob (223 KB, L2-resident) into LDS, coalesced uint4 ----
  {
    const uint4* src = (const uint4*)wsblob;
    uint4* dst = (uint4*)smem;
    for (int i = threadIdx.x; i < (int)(WBLOB / 16u); i += blockDim.x)
      dst[i] = src[i];
  }
  __syncthreads();

  const int lane = threadIdx.x & 31;
  const int wave = threadIdx.x >> 5;
  const int am = lane & 15;        // A layout: lane -> M row
  const int ag = lane >> 4;        // A layout: half-wave K group
  const int cn = lane & 15;        // C layout: lane -> N col
  const int cm0 = ag << 3;         // C layout: +8 rows for lanes 16..31

  const int tile = blockIdx.x * WAVES + wave;      // 0..1023
  const int b0 = tile * 16;
  const float* xrow = x + (size_t)(b0 + am) * TSTEPS * NNODES;

  unsigned short* xst = &xmix_st[wave][0];
  unsigned short* hst = &h_st[wave][0];

  unsigned Ah[8], AhR[8];
  float c0[2][8];
#pragma unroll
  for (int v = 0; v < 8; ++v) {
    Ah[v] = 0u; AhR[v] = 0u; c0[0][v] = 0.f; c0[1][v] = 0.f;
  }

  const unsigned short* attn = (const unsigned short*)(smem + OFF_ATTN_R);
  const unsigned short* wih  = (const unsigned short*)(smem + OFF_WIH_R);
  const unsigned short* whh  = (const unsigned short*)(smem + OFF_WHH_R);
  const float*          bias = (const float*)(smem + OFF_BIAS_R);

#pragma unroll 1
  for (int t = 0; t < TSTEPS; ++t) {
    if (t == 4) {                                  // switch recent -> period
#pragma unroll
      for (int v = 0; v < 8; ++v) {
        AhR[v] = Ah[v]; Ah[v] = 0u; c0[0][v] = 0.f; c0[1][v] = 0.f;
      }
      attn = (const unsigned short*)(smem + OFF_ATTN_P);
      wih  = (const unsigned short*)(smem + OFF_WIH_P);
      whh  = (const unsigned short*)(smem + OFF_WHH_P);
      bias = (const float*)(smem + OFF_BIAS_P);
    }

    // -- x tile (16x156 f32) -> WMMA A-layout bf16 registers, K padded 160 --
    ABf Ax[5];
    const float* xr = xrow + (size_t)t * NNODES;
#pragma unroll
    for (int kk = 0; kk < 5; ++kk) {
#pragma unroll
      for (int v = 0; v < 8; ++v) {
        int k = kk * 32 + a_k(v, ag);
        unsigned pk = 0u;
        if (k < NNODES) {                          // pair (k,k+1) both valid
          float2 f2 = *(const float2*)(xr + k);
          pk = pack2_bf16(f2.x, f2.y);
        }
        Ax[kk].u[v] = pk;
      }
    }

    // -- xmix = x @ attn^T : 10 N-tiles x 5 K-steps; bf16 results to LDS --
#pragma unroll
    for (int n = 0; n < 10; ++n) {
      v8f acc;
#pragma unroll
      for (int v = 0; v < 8; ++v) acc[v] = 0.f;
#pragma unroll
      for (int kk = 0; kk < 5; ++kk) {
        ABf Bt = load_B(attn, NP, n * 16, kk * 32, lane);
        acc = wmma_bf16(Ax[kk], Bt, acc);
      }
#pragma unroll
      for (int v = 0; v < 8; ++v)
        xst[(v + cm0) * NP + n * 16 + cn] = f2bf_h(acc[v]);
    }

    // -- re-read xmix in A layout (same-wave DS ops stay ordered) --
    ABf Axm[5];
    const unsigned short* xsp = xst + am * NP;
#pragma unroll
    for (int kk = 0; kk < 5; ++kk) {
#pragma unroll
      for (int v = 0; v < 8; ++v)
        Axm[kk].u[v] = ld_pair(xsp + kk * 32 + a_k(v, ag));
    }

    // -- gates g = xm @ Wih^T + h @ Whh^T + bias : 8 N-tiles of 16 --
    ABf AhV;
#pragma unroll
    for (int v = 0; v < 8; ++v) AhV.u[v] = Ah[v];
    v8f g[8];
#pragma unroll
    for (int n = 0; n < 8; ++n) {
      float bval = bias[n * 16 + cn];
      v8f acc;
#pragma unroll
      for (int v = 0; v < 8; ++v) acc[v] = bval;
#pragma unroll
      for (int kk = 0; kk < 5; ++kk) {
        ABf Bt = load_B(wih, NP, n * 16, kk * 32, lane);
        acc = wmma_bf16(Axm[kk], Bt, acc);
      }
      ABf Bh = load_B(whh, HL, n * 16, 0, lane);   // recurrent, K=32
      acc = wmma_bf16(AhV, Bh, acc);
      g[n] = acc;
    }

    // -- LSTM cell elementwise; h -> LDS -> A layout for next step --
#pragma unroll
    for (int tt = 0; tt < 2; ++tt) {
#pragma unroll
      for (int v = 0; v < 8; ++v) {
        float iv = sigm(g[0 + tt][v]);
        float fv = sigm(g[2 + tt][v]);
        float gv = tanhf(g[4 + tt][v]);
        float ov = sigm(g[6 + tt][v]);
        float cc = fv * c0[tt][v] + iv * gv;
        c0[tt][v] = cc;
        hst[(v + cm0) * HL + tt * 16 + cn] = f2bf_h(ov * tanhf(cc));
      }
    }
    {
      const unsigned short* hp = hst + am * HL;
#pragma unroll
      for (int v = 0; v < 8; ++v) Ah[v] = ld_pair(hp + a_k(v, ag));
    }
  }

  // -- fc: out = [hr | hp] @ fcW^T + fcb  (K = 64 = two A tiles) --
  ABf Ahr, Ahp;
#pragma unroll
  for (int v = 0; v < 8; ++v) { Ahr.u[v] = AhR[v]; Ahp.u[v] = Ah[v]; }
  const unsigned short* fcw = (const unsigned short*)(smem + OFF_FCW);
  const float* fcb = (const float*)(smem + OFF_FCB);
#pragma unroll
  for (int n = 0; n < 10; ++n) {
    float bval = fcb[n * 16 + cn];
    v8f acc;
#pragma unroll
    for (int v = 0; v < 8; ++v) acc[v] = bval;
    ABf B0 = load_B(fcw, 2 * HL, n * 16, 0, lane);
    ABf B1 = load_B(fcw, 2 * HL, n * 16, 32, lane);
    acc = wmma_bf16(Ahr, B0, acc);
    acc = wmma_bf16(Ahp, B1, acc);
    int col = n * 16 + cn;
    if (col < NNODES) {
#pragma unroll
      for (int v = 0; v < 8; ++v)
        out[(size_t)(b0 + v + cm0) * NNODES + col] = acc[v];
    }
  }
}

// ---------------------------------------------------------------------------
extern "C" void kernel_launch(void* const* d_in, const int* in_sizes, int n_in,
                              void* d_out, int out_size, void* d_ws,
                              size_t ws_size, hipStream_t stream) {
  const float* emb   = (const float*)d_in[0];
  const float* x     = (const float*)d_in[1];
  const int*   adj   = (const int*)d_in[2];
  const float* Wr    = (const float*)d_in[3];
  const float* ar    = (const float*)d_in[4];
  const float* Wp    = (const float*)d_in[5];
  const float* ap    = (const float*)d_in[6];
  const float* wih_r = (const float*)d_in[7];
  const float* whh_r = (const float*)d_in[8];
  const float* bih_r = (const float*)d_in[9];
  const float* bhh_r = (const float*)d_in[10];
  const float* wih_p = (const float*)d_in[11];
  const float* whh_p = (const float*)d_in[12];
  const float* bih_p = (const float*)d_in[13];
  const float* bhh_p = (const float*)d_in[14];
  const float* fcW   = (const float*)d_in[15];
  const float* fcB   = (const float*)d_in[16];

  unsigned char* ws = (unsigned char*)d_ws;
  float* sbuf = (float*)(ws + OFF_SBUF);

  gat_scores_kernel<<<dim3(NNODES, 2), 128, 0, stream>>>(emb, Wr, ar, Wp, ap,
                                                         sbuf);
  gat_softmax_kernel<<<dim3(NNODES, 2), 256, 0, stream>>>(adj, sbuf, ws);
  prep_weights_kernel<<<dim3(80, 9), 256, 0, stream>>>(
      wih_r, whh_r, bih_r, bhh_r, wih_p, whh_p, bih_p, bhh_p, fcW, fcB, ws);

  // Opt into >64KB dynamic LDS for the weight image (WGP has 320 KB).
  (void)hipFuncSetAttribute(
      reinterpret_cast<const void*>(&fused_gat_lstm_kernel),
      hipFuncAttributeMaxDynamicSharedMemorySize, (int)WBLOB);

  fused_gat_lstm_kernel<<<BATCH / (16 * WAVES), 32 * WAVES, WBLOB, stream>>>(
      x, ws, (float*)d_out);
}
